// Attention_33578054320502
// MI455X (gfx1250) — compile-verified, matrix operand order checked by reading, not run on hardware
//
#include <hip/hip_runtime.h>
#include <hip/hip_bf16.h>

#define BATCH  4
#define SEQ    2048
#define DMODEL 1024
#define DKV    1024
#define EDIM   3072   // 2*D_K + D_V
#define WAVES  4
#define KTILE  32

typedef __attribute__((ext_vector_type(16))) __bf16       v16bf;
typedef __attribute__((ext_vector_type(8)))  float        v8f;
typedef __attribute__((ext_vector_type(4)))  unsigned int u32x4;
typedef __attribute__((ext_vector_type(4)))  int          i32x4;

union FragBF {
    v16bf v;
    u32x4 q[2];   // 2 x 16B = 16 bf16
};

__device__ inline unsigned short f32_to_bf16_bits(float f) {
    union { float f; unsigned u; } x; x.f = f;
    unsigned r = x.u + 0x7FFFu + ((x.u >> 16) & 1u);  // round-to-nearest-even
    return (unsigned short)(r >> 16);
}

// ---- CDNA5 async global->LDS copy (ASYNCcnt path), with sync fallback ----
#if defined(__AMDGCN__) && __has_builtin(__builtin_amdgcn_global_load_async_to_lds_b128) && __has_builtin(__builtin_amdgcn_s_wait_asynccnt)
#define HAVE_ASYNC_LDS 1
#else
#define HAVE_ASYNC_LDS 0
#endif

__device__ inline void async_copy16(const unsigned short* g, unsigned short* l) {
#if HAVE_ASYNC_LDS
    __builtin_amdgcn_global_load_async_to_lds_b128(
        (__attribute__((address_space(1))) i32x4*)(g),
        (__attribute__((address_space(3))) i32x4*)(l), 0, 0);
#else
    *(u32x4*)l = *(const u32x4*)g;
#endif
}

__device__ inline void async_wait_all() {
#if HAVE_ASYNC_LDS
    __builtin_amdgcn_s_wait_asynccnt(0);
#endif
}

// ---------------------------------------------------------------- kernel 1
__global__ void cvt_f32_bf16(const float* __restrict__ src,
                             unsigned short* __restrict__ dst, int n) {
    int i = blockIdx.x * blockDim.x + threadIdx.x;
    int stride = gridDim.x * blockDim.x;
    for (; i < n; i += stride) dst[i] = f32_to_bf16_bits(src[i]);
}

// ---------------------------------------------------------------- kernel 2
// qkv[b,n,e] = sum_d t[b,n,d]*W[e,d] + bias[e]
// 64x64 output block per 4-wave workgroup; LDS-staged, double-buffered,
// async global->LDS copies; wave w owns rows [w*16, w*16+16) x 64 cols.
__global__ __launch_bounds__(128)
void qkv_gemm(const unsigned short* __restrict__ tb,
              const unsigned short* __restrict__ Wb,
              const float* __restrict__ bias,
              unsigned short* __restrict__ Qb,
              unsigned short* __restrict__ Kb,
              unsigned short* __restrict__ Vt) {
    __shared__ unsigned short As[2][64 * KTILE];   // 4 KB each buffer
    __shared__ unsigned short Bs[2][64 * KTILE];

    const int tid  = threadIdx.x;
    const int lane = tid & 31;
    const int wave = tid >> 5;
    const int r    = lane & 15;   // A: row | B: out column | C: column
    const int half = lane >> 4;   // K-group select

    const int EB = EDIM / 64;     // 48
    const int MB = SEQ / 64;      // 32
    const int eb = blockIdx.x % EB;
    const int mb = (blockIdx.x / EB) % MB;
    const int b  = blockIdx.x / (EB * MB);

    const unsigned short* Abase = tb + ((size_t)b * SEQ + mb * 64) * DMODEL;
    const unsigned short* Bbase = Wb + (size_t)(eb * 64) * DMODEL;

    // cooperative tile fetch: 64 rows x 32 bf16 = 256 16B-chunks per tile;
    // 128 threads -> 2 chunks of A and 2 of B each.
    auto issue = [&](int buf, int k0) {
        for (int i = 0; i < 2; ++i) {
            const int ch  = tid + i * 128;
            const int row = ch >> 2;
            const int c8  = (ch & 3) * 8;
            async_copy16(Abase + (size_t)row * DMODEL + k0 + c8, &As[buf][row * KTILE + c8]);
            async_copy16(Bbase + (size_t)row * DMODEL + k0 + c8, &Bs[buf][row * KTILE + c8]);
        }
    };

    v8f acc[4] = {};
    issue(0, 0);
    const int NK = DMODEL / KTILE;   // 32
    for (int kb = 0; kb < NK; ++kb) {
        async_wait_all();
        __syncthreads();
        if (kb + 1 < NK) issue((kb + 1) & 1, (kb + 1) * KTILE);

        const unsigned short* Ar = &As[kb & 1][(wave * 16 + r) * KTILE + half * 8];
        FragBF a;
        a.q[0] = *(const u32x4*)(Ar);
        a.q[1] = *(const u32x4*)(Ar + 16);
        for (int j = 0; j < 4; ++j) {
            const unsigned short* Br = &Bs[kb & 1][(j * 16 + r) * KTILE + half * 8];
            FragBF w;
            w.q[0] = *(const u32x4*)(Br);
            w.q[1] = *(const u32x4*)(Br + 16);
            acc[j] = __builtin_amdgcn_wmma_f32_16x16x32_bf16(false, a.v, false, w.v,
                                                             (short)0, acc[j], false, false);
        }
    }

    const int e0 = eb * 64;
    const int m0 = mb * 64 + wave * 16 + half * 8;   // first row held by this lane
    for (int j = 0; j < 4; ++j) {
        const int ecol = e0 + j * 16 + r;
        const float bv = bias[ecol];
        if (ecol < DKV) {                       // Q, pre-scaled by 1/sqrt(D_K)
            for (int c = 0; c < 8; ++c)
                Qb[((size_t)b * SEQ + m0 + c) * DKV + ecol] =
                    f32_to_bf16_bits((acc[j][c] + bv) * 0.03125f);
        } else if (ecol < 2 * DKV) {            // K
            for (int c = 0; c < 8; ++c)
                Kb[((size_t)b * SEQ + m0 + c) * DKV + (ecol - DKV)] =
                    f32_to_bf16_bits(acc[j][c] + bv);
        } else {                                // V stored transposed: Vt[b][v][n]
            const int v = ecol - 2 * DKV;
            for (int c = 0; c < 8; ++c)
                Vt[((size_t)b * DKV + v) * SEQ + (m0 + c)] =
                    f32_to_bf16_bits(acc[j][c] + bv);
        }
    }
}

// ---------------------------------------------------------------- kernel 3
// Fused attention for one (b, 16-row q-tile). 4 waves per block.
// Q tile staged in LDS (async) and shared by all waves; K/Vt streamed.
__global__ __launch_bounds__(128)
void attn(const unsigned short* __restrict__ Qb,
          const unsigned short* __restrict__ Kb,
          const unsigned short* __restrict__ Vt,
          float* __restrict__ out) {
    extern __shared__ char smem_raw[];
    unsigned short* P  = (unsigned short*)smem_raw;                    // 16 x SEQ bf16 (64 KB)
    unsigned short* Qs = (unsigned short*)(smem_raw + 16 * SEQ * 2);   // 16 x DKV bf16 (32 KB)
    float* m_part = (float*)(smem_raw + 16 * SEQ * 2 + 16 * DKV * 2);  // [WAVES][16]
    float* l_part = m_part + WAVES * 16;                               // [WAVES][16]

    const int tid  = threadIdx.x;
    const int lane = tid & 31;
    const int wave = tid >> 5;
    const int r    = lane & 15;
    const int half = lane >> 4;

    const int mt = blockIdx.x % (SEQ / 16);
    const int b  = blockIdx.x / (SEQ / 16);

    // ---- stage the 16 x DKV Q tile into LDS (shared across waves/passes)
    const unsigned short* Qg = Qb + ((size_t)b * SEQ + mt * 16) * DKV;
    for (int ch = tid; ch < 16 * DKV / 8; ch += 128)
        async_copy16(Qg + ch * 8, Qs + ch * 8);
    async_wait_all();
    __syncthreads();

    const unsigned short* qrow = Qs + (size_t)r * DKV + half * 8;

    float mrow[8], lrow[8];
    for (int c = 0; c < 8; ++c) { mrow[c] = -3.0e38f; lrow[c] = 0.f; }

    // ---- pass 1: online row stats over key tiles owned by this wave
    for (int ct = wave; ct < SEQ / 16; ct += WAVES) {
        const unsigned short* krow =
            Kb + ((size_t)b * SEQ + ct * 16 + r) * DKV + half * 8;
        v8f acc = {};
        for (int k0 = 0; k0 < DKV; k0 += 32) {
            FragBF a, kk;
            a.q[0]  = *(const u32x4*)(qrow + k0);
            a.q[1]  = *(const u32x4*)(qrow + k0 + 16);
            kk.q[0] = *(const u32x4*)(krow + k0);
            kk.q[1] = *(const u32x4*)(krow + k0 + 16);
            acc = __builtin_amdgcn_wmma_f32_16x16x32_bf16(false, a.v, false, kk.v,
                                                          (short)0, acc, false, false);
        }
        for (int c = 0; c < 8; ++c) {
            float s = acc[c];
            float t = s;
            for (int off = 1; off < 16; off <<= 1) t = fmaxf(t, __shfl_xor(t, off, 16));
            float nm = fmaxf(mrow[c], t);
            float e  = __expf(s - nm);
            for (int off = 1; off < 16; off <<= 1) e += __shfl_xor(e, off, 16);
            lrow[c] = lrow[c] * __expf(mrow[c] - nm) + e;
            mrow[c] = nm;
        }
    }

    // ---- merge stats across waves via LDS
    if (r == 0)
        for (int c = 0; c < 8; ++c) {
            m_part[wave * 16 + half * 8 + c] = mrow[c];
            l_part[wave * 16 + half * 8 + c] = lrow[c];
        }
    __syncthreads();

    float mg[8], lg[8];
    for (int c = 0; c < 8; ++c) {
        const int row = half * 8 + c;
        float m = m_part[row];
        for (int w = 1; w < WAVES; ++w) m = fmaxf(m, m_part[w * 16 + row]);
        float l = 0.f;
        for (int w = 0; w < WAVES; ++w)
            l += l_part[w * 16 + row] * __expf(m_part[w * 16 + row] - m);
        mg[c] = m; lg[c] = l;
    }

    // ---- pass 2: recompute S, write P = exp(S - mg) to LDS as bf16
    for (int ct = wave; ct < SEQ / 16; ct += WAVES) {
        const unsigned short* krow =
            Kb + ((size_t)b * SEQ + ct * 16 + r) * DKV + half * 8;
        v8f acc = {};
        for (int k0 = 0; k0 < DKV; k0 += 32) {
            FragBF a, kk;
            a.q[0]  = *(const u32x4*)(qrow + k0);
            a.q[1]  = *(const u32x4*)(qrow + k0 + 16);
            kk.q[0] = *(const u32x4*)(krow + k0);
            kk.q[1] = *(const u32x4*)(krow + k0 + 16);
            acc = __builtin_amdgcn_wmma_f32_16x16x32_bf16(false, a.v, false, kk.v,
                                                          (short)0, acc, false, false);
        }
        for (int c = 0; c < 8; ++c)
            P[(size_t)(half * 8 + c) * SEQ + ct * 16 + r] =
                f32_to_bf16_bits(__expf(acc[c] - mg[c]));
    }
    __syncthreads();

    // ---- phase 3: O = P * V  (A-fragments from LDS, B from Vt rows)
    const unsigned short* prow = P + (size_t)r * SEQ + half * 8;
    for (int vt = wave; vt < DKV / 16; vt += WAVES) {
        const unsigned short* vrow =
            Vt + ((size_t)b * DKV + vt * 16 + r) * SEQ + half * 8;
        v8f acc = {};
        for (int k0 = 0; k0 < SEQ; k0 += 32) {
            FragBF a, vv;
            a.q[0]  = *(const u32x4*)(prow + k0);
            a.q[1]  = *(const u32x4*)(prow + k0 + 16);
            vv.q[0] = *(const u32x4*)(vrow + k0);
            vv.q[1] = *(const u32x4*)(vrow + k0 + 16);
            acc = __builtin_amdgcn_wmma_f32_16x16x32_bf16(false, a.v, false, vv.v,
                                                          (short)0, acc, false, false);
        }
        for (int c = 0; c < 8; ++c) {
            const int row = half * 8 + c;
            out[((size_t)b * SEQ + mt * 16 + row) * DKV + vt * 16 + r] = acc[c] / lg[c];
        }
    }
}

// ---------------------------------------------------------------- launcher
extern "C" void kernel_launch(void* const* d_in, const int* in_sizes, int n_in,
                              void* d_out, int out_size, void* d_ws, size_t ws_size,
                              hipStream_t stream) {
    const float* t    = (const float*)d_in[0];
    const float* W    = (const float*)d_in[1];
    const float* bias = (const float*)d_in[2];
    float*       out  = (float*)d_out;

    // workspace layout (bf16 arrays as unsigned short)
    char* ws = (char*)d_ws;
    const size_t nT = (size_t)BATCH * SEQ * DMODEL;   // 8M elems
    const size_t nW = (size_t)EDIM * DMODEL;          // 3M elems
    const size_t nQ = (size_t)BATCH * SEQ * DKV;      // 8M elems
    unsigned short* tb = (unsigned short*)(ws);
    unsigned short* Wb = (unsigned short*)(ws + nT * 2);
    unsigned short* Qb = (unsigned short*)(ws + (nT + nW) * 2);
    unsigned short* Kb = (unsigned short*)(ws + (nT + nW + nQ) * 2);
    unsigned short* Vt = (unsigned short*)(ws + (nT + nW + 2 * nQ) * 2);

    // 1) convert inputs to bf16
    cvt_f32_bf16<<<4096, 256, 0, stream>>>(t, tb, (int)nT);
    cvt_f32_bf16<<<2048, 256, 0, stream>>>(W, Wb, (int)nW);

    // 2) QKV projection: 64x64 blocks, 4 waves/block
    const int blocks2 = BATCH * (SEQ / 64) * (EDIM / 64);
    qkv_gemm<<<blocks2, 32 * WAVES, 0, stream>>>(tb, Wb, bias, Qb, Kb, Vt);

    // 3) fused attention: one block per (b, 16-row q tile)
    const size_t smem = 16 * SEQ * 2 + 16 * DKV * 2 + 2 * WAVES * 16 * sizeof(float);
    attn<<<BATCH * (SEQ / 16), 32 * WAVES, smem, stream>>>(Qb, Kb, Vt, out);
}
